// LSTMModel_38749194944762
// MI455X (gfx1250) — matrix operand order repaired and checked by reading, not compiled, not run on hardware
//
#include <hip/hip_runtime.h>

// ---------------------------------------------------------------------------
// Persistent 2-layer LSTM for MI455X (gfx1250, wave32, WMMA f16->f32).
//   B=256, T=512, H=512.  64 persistent WGs x 128 threads (4 waves).
//   WG (p, j) owns batch rows [p*128, p*128+128) and hidden slice
//   [j*16, j*16+16): computes gates i,f,g,o for that slice.
//   Weights (fp16, transposed to [n][k]) live in LDS (194KB/WGP of 320KB).
//   c-state lives in registers for the whole sequence. h-state double-
//   buffered fp16 in workspace (L2-resident). ONE grid barrier per step:
//     { L1(t); fence; bar; L2(t); }  -- double-buffering makes the
//   L2(t) / L1(t+1) overlap race-free (see analysis).
// ---------------------------------------------------------------------------

typedef __attribute__((ext_vector_type(16))) _Float16 v16h;
typedef __attribute__((ext_vector_type(8)))  _Float16 v8h;
typedef __attribute__((ext_vector_type(8)))  float    v8f;

#define HID   512
#define BATCH 256
#define TLEN  512
#define NG    2048              // 4*HID gate columns
#define JW    16                // hidden units per workgroup
#define MPART 2                 // batch partitions
#define NWGS  ((HID / JW) * MPART)   // 64 workgroups
#define THREADS 128             // 4 waves
#define K1    512
#define K2    1024
#define K1P   (K1 + 8)          // pad 8 halves (16B) -> bank-conflict-free B reads
#define K2P   (K2 + 8)
#define LDS_BYTES ((64 * K1P + 64 * K2P) * 2)   // 198,656 B < 320 KB/WGP

union V16 { v16h v; v8h h[2]; };

__device__ __forceinline__ float sigf(float x)     { return 1.0f / (1.0f + __expf(-x)); }
__device__ __forceinline__ float tanhfast(float x) { return 2.0f / (1.0f + __expf(-2.0f * x)) - 1.0f; }

__device__ __forceinline__ void grid_barrier(unsigned* cnt, unsigned* gen, unsigned nb) {
    __syncthreads();
    if (threadIdx.x == 0) {
        __threadfence();
        unsigned g = atomicAdd(gen, 0u);
        unsigned arrived = atomicAdd(cnt, 1u);
        if (arrived == nb - 1u) {
            atomicExch(cnt, 0u);
            __threadfence();
            atomicAdd(gen, 1u);
        } else {
            while (atomicAdd(gen, 0u) == g) { __builtin_amdgcn_s_sleep(1); }
        }
        __threadfence();
    }
    __syncthreads();
}

// ---- one-time per-launch init: zero h buffers, fold biases, reset barrier --
__global__ void init_state_kernel(_Float16* __restrict__ h1, _Float16* __restrict__ h2,
                                  float* __restrict__ b1, float* __restrict__ b2,
                                  const float* __restrict__ b_ih1, const float* __restrict__ b_hh1,
                                  const float* __restrict__ b_ih2, const float* __restrict__ b_hh2,
                                  unsigned* __restrict__ bar) {
    const int idx    = blockIdx.x * blockDim.x + threadIdx.x;
    const int stride = gridDim.x * blockDim.x;
    const int totalh = 2 * BATCH * HID;   // halves per double-buffered state
    for (int i = idx; i < totalh; i += stride) {
        h1[i] = (_Float16)0.0f;
        h2[i] = (_Float16)0.0f;
    }
    for (int i = idx; i < NG; i += stride) {
        b1[i] = b_ih1[i] + b_hh1[i];
        b2[i] = b_ih2[i] + b_hh2[i];
    }
    if (idx == 0) { bar[0] = 0u; bar[32] = 0u; }
}

// ---- one-time per-launch: fp32 -> fp16 transposed weights [n][k] -----------
__global__ void convert_weights_kernel(const float* __restrict__ w_hh1,
                                       const float* __restrict__ w_ih2,
                                       const float* __restrict__ w_hh2,
                                       _Float16* __restrict__ wt1,
                                       _Float16* __restrict__ wt2) {
    const int idx    = blockIdx.x * blockDim.x + threadIdx.x;
    const int stride = gridDim.x * blockDim.x;
    const int total1 = NG * K1;
    const int total2 = NG * K2;
    for (int i = idx; i < total1; i += stride) {
        const int n = i / K1, k = i % K1;
        wt1[i] = (_Float16)w_hh1[(size_t)k * NG + n];
    }
    for (int i = idx; i < total2; i += stride) {
        const int n = i / K2, k = i % K2;
        wt2[i] = (_Float16)((k < K1) ? w_ih2[(size_t)k * NG + n]
                                     : w_hh2[(size_t)(k - K1) * NG + n]);
    }
}

// ---------------------------- main persistent kernel ------------------------
__global__ __launch_bounds__(THREADS, 1)
void lstm_persistent(const float* __restrict__ x,        // [B,T,2]
                     const float* __restrict__ w_ih1,    // [2,2048]
                     const float* __restrict__ b1,       // folded [2048]
                     const float* __restrict__ b2,       // folded [2048]
                     const _Float16* __restrict__ wt1,   // [2048][512]  (n-major)
                     const _Float16* __restrict__ wt2,   // [2048][1024] (n-major)
                     _Float16* __restrict__ h1buf,       // 2 x [B][H] fp16
                     _Float16* __restrict__ h2buf,       // 2 x [B][H] fp16
                     unsigned* __restrict__ bar) {
    extern __shared__ _Float16 lds[];
    _Float16* w1s = lds;                 // [64][K1P]  column-major (k contiguous)
    _Float16* w2s = lds + 64 * K1P;      // [64][K2P]

    const int tid   = threadIdx.x;
    const int lane  = tid & 31;
    const int wv    = tid >> 5;                  // wave 0..3
    const int slice = blockIdx.x & 31;           // hidden slice 0..31
    const int mpart = blockIdx.x >> 5;           // batch partition 0..1
    const int j0    = slice * JW;                // hidden slice base

    // ---- stage this WG's weight slices into LDS (once) ----
    for (int n = wv; n < 64; n += 4) {
        const int ng = (n >> 4) * HID + j0 + (n & 15);   // gate g = n/16, unit = n%16
        {
            const _Float16* src = wt1 + (size_t)ng * K1;
            _Float16* dst = w1s + n * K1P;
            for (int k = lane * 8; k < K1; k += 256)
                *(v8h*)(dst + k) = *(const v8h*)(src + k);
        }
        {
            const _Float16* src = wt2 + (size_t)ng * K2;
            _Float16* dst = w2s + n * K2P;
            for (int k = lane * 8; k < K2; k += 256)
                *(v8h*)(dst + k) = *(const v8h*)(src + k);
        }
    }
    __syncthreads();

    // WMMA lane geometry (16x16x32 f16, wave32):
    //   A: lane holds row (lane&15); K-halves: lanes 0-15 -> K {0..7,16..23},
    //      lanes 16-31 -> K {8..15,24..31}  => two 16B loads per tile.
    //   B: lane holds col (lane&15); same K-halves as A => two 16B loads per tile.
    //   D: vgpr r, lane l -> row r + 8*(l>>4), col l&15.
    const int nl    = lane & 15;
    const int hi    = lane >> 4;
    const int khalf = hi * 8;
    const int jg    = j0 + nl;                   // global hidden unit this lane owns
    const int mrow  = mpart * 128 + wv * 32;     // wave's 32 batch rows (2 M-tiles)

    // hoisted per-lane constants: input projection weights + folded biases
    float w0g[4], w1g[4], bb1[4], bb2[4];
#pragma unroll
    for (int g = 0; g < 4; ++g) {
        const int ng = g * HID + jg;
        w0g[g] = w_ih1[ng];
        w1g[g] = w_ih1[NG + ng];
        bb1[g] = b1[ng];
        bb2[g] = b2[ng];
    }

    // cell state lives in registers for the whole sequence
    float c1r[16], c2r[16];
#pragma unroll
    for (int i = 0; i < 16; ++i) { c1r[i] = 0.0f; c2r[i] = 0.0f; }

    const v8f vzero = {0.f, 0.f, 0.f, 0.f, 0.f, 0.f, 0.f, 0.f};

    for (int t = 0; t < TLEN; ++t) {
        const int rb = t & 1;            // read h(t-1)
        const int wb = rb ^ 1;           // write h(t)
        const _Float16* h1p = h1buf + (size_t)rb * (BATCH * HID);
        const _Float16* h2p = h2buf + (size_t)rb * (BATCH * HID);
        _Float16* h1o = h1buf + (size_t)wb * (BATCH * HID);
        _Float16* h2o = h2buf + (size_t)wb * (BATCH * HID);

        v8f acc[2][4];

        // ================= layer 1: gates1 = h1(t-1) @ w_hh1 =================
#pragma unroll
        for (int mt = 0; mt < 2; ++mt)
#pragma unroll
            for (int g = 0; g < 4; ++g) acc[mt][g] = vzero;

        for (int kt = 0; kt < K1 / 32; ++kt) {
            const int kb = kt * 32;
            V16 a0, a1;
            {
                const _Float16* ap = h1p + (size_t)(mrow + nl) * HID + kb + khalf;
                a0.h[0] = *(const v8h*)ap;
                a0.h[1] = *(const v8h*)(ap + 16);
            }
            {
                const _Float16* ap = h1p + (size_t)(mrow + 16 + nl) * HID + kb + khalf;
                a1.h[0] = *(const v8h*)ap;
                a1.h[1] = *(const v8h*)(ap + 16);
            }
#pragma unroll
            for (int g = 0; g < 4; ++g) {
                V16 bt;
                const _Float16* bp = w1s + (g * 16 + nl) * K1P + kb + hi * 8;
                bt.h[0] = *(const v8h*)bp;
                bt.h[1] = *(const v8h*)(bp + 16);
                acc[0][g] = __builtin_amdgcn_wmma_f32_16x16x32_f16(
                    false, a0.v, false, bt.v, (short)0, acc[0][g], false, false);
                acc[1][g] = __builtin_amdgcn_wmma_f32_16x16x32_f16(
                    false, a1.v, false, bt.v, (short)0, acc[1][g], false, false);
            }
        }

        // nonlinearity + x-projection (K=2 done scalar) + h1 write
#pragma unroll
        for (int mt = 0; mt < 2; ++mt) {
#pragma unroll
            for (int r = 0; r < 8; ++r) {
                const int row = mrow + mt * 16 + r + 8 * hi;
                const float2 xv = *(const float2*)(x + (size_t)row * (TLEN * 2) + 2 * t);
                const float gi = acc[mt][0][r] + xv.x * w0g[0] + xv.y * w1g[0] + bb1[0];
                const float gf = acc[mt][1][r] + xv.x * w0g[1] + xv.y * w1g[1] + bb1[1];
                const float gg = acc[mt][2][r] + xv.x * w0g[2] + xv.y * w1g[2] + bb1[2];
                const float go = acc[mt][3][r] + xv.x * w0g[3] + xv.y * w1g[3] + bb1[3];
                const float c  = sigf(gf) * c1r[mt * 8 + r] + sigf(gi) * tanhfast(gg);
                c1r[mt * 8 + r] = c;
                h1o[(size_t)row * HID + jg] = (_Float16)(sigf(go) * tanhfast(c));
            }
        }

        // single grid sync per timestep: h1(t) complete & visible device-wide.
        // (h2(t-1) writes from L2(t-1) are also covered by this barrier.)
        __threadfence();
        grid_barrier(bar, bar + 32, (unsigned)gridDim.x);

        // ====== layer 2: gates2 = [h1(t); h2(t-1)] @ [w_ih2; w_hh2] =========
#pragma unroll
        for (int mt = 0; mt < 2; ++mt)
#pragma unroll
            for (int g = 0; g < 4; ++g) acc[mt][g] = vzero;

        for (int kt = 0; kt < K2 / 32; ++kt) {
            const _Float16* hp = (kt < 16) ? h1o : h2p;
            const int kb = (kt < 16 ? kt : kt - 16) * 32;
            V16 a0, a1;
            {
                const _Float16* ap = hp + (size_t)(mrow + nl) * HID + kb + khalf;
                a0.h[0] = *(const v8h*)ap;
                a0.h[1] = *(const v8h*)(ap + 16);
            }
            {
                const _Float16* ap = hp + (size_t)(mrow + 16 + nl) * HID + kb + khalf;
                a1.h[0] = *(const v8h*)ap;
                a1.h[1] = *(const v8h*)(ap + 16);
            }
            const int kw = kt * 32;
#pragma unroll
            for (int g = 0; g < 4; ++g) {
                V16 bt;
                const _Float16* bp = w2s + (g * 16 + nl) * K2P + kw + hi * 8;
                bt.h[0] = *(const v8h*)bp;
                bt.h[1] = *(const v8h*)(bp + 16);
                acc[0][g] = __builtin_amdgcn_wmma_f32_16x16x32_f16(
                    false, a0.v, false, bt.v, (short)0, acc[0][g], false, false);
                acc[1][g] = __builtin_amdgcn_wmma_f32_16x16x32_f16(
                    false, a1.v, false, bt.v, (short)0, acc[1][g], false, false);
            }
        }

#pragma unroll
        for (int mt = 0; mt < 2; ++mt) {
#pragma unroll
            for (int r = 0; r < 8; ++r) {
                const int row = mrow + mt * 16 + r + 8 * hi;
                const float gi = acc[mt][0][r] + bb2[0];
                const float gf = acc[mt][1][r] + bb2[1];
                const float gg = acc[mt][2][r] + bb2[2];
                const float go = acc[mt][3][r] + bb2[3];
                const float c  = sigf(gf) * c2r[mt * 8 + r] + sigf(gi) * tanhfast(gg);
                c2r[mt * 8 + r] = c;
                h2o[(size_t)row * HID + jg] = (_Float16)(sigf(go) * tanhfast(c));
            }
        }
        // no trailing barrier: next iteration's L1 doesn't touch h2, and the
        // h1 buffer it overwrites has no remaining readers (see header note).
    }
}

// ---- tail: logits = h2 @ fc_w + fc_b ; softmax (tiny) ----------------------
__global__ void fc_softmax_kernel(const _Float16* __restrict__ h2,
                                  const float* __restrict__ fc_w,  // [512][11]
                                  const float* __restrict__ fc_b,  // [11]
                                  float* __restrict__ out) {       // [256][11]
    const int row = blockIdx.x * blockDim.x + threadIdx.x;
    if (row >= BATCH) return;
    float lo[11];
#pragma unroll
    for (int o = 0; o < 11; ++o) lo[o] = fc_b[o];
    for (int k = 0; k < HID; ++k) {
        const float hv = (float)h2[(size_t)row * HID + k];
#pragma unroll
        for (int o = 0; o < 11; ++o) lo[o] += hv * fc_w[k * 11 + o];
    }
    float m = lo[0];
#pragma unroll
    for (int o = 1; o < 11; ++o) m = fmaxf(m, lo[o]);
    float s = 0.0f;
#pragma unroll
    for (int o = 0; o < 11; ++o) { lo[o] = __expf(lo[o] - m); s += lo[o]; }
    const float inv = 1.0f / s;
#pragma unroll
    for (int o = 0; o < 11; ++o) out[row * 11 + o] = lo[o] * inv;
}

// ---------------------------------------------------------------------------
extern "C" void kernel_launch(void* const* d_in, const int* in_sizes, int n_in,
                              void* d_out, int out_size, void* d_ws, size_t ws_size,
                              hipStream_t stream) {
    const float* x     = (const float*)d_in[0];
    const float* w_ih1 = (const float*)d_in[1];
    const float* w_hh1 = (const float*)d_in[2];
    const float* b_ih1 = (const float*)d_in[3];
    const float* b_hh1 = (const float*)d_in[4];
    const float* w_ih2 = (const float*)d_in[5];
    const float* w_hh2 = (const float*)d_in[6];
    const float* b_ih2 = (const float*)d_in[7];
    const float* b_hh2 = (const float*)d_in[8];
    const float* fc_w  = (const float*)d_in[9];
    const float* fc_b  = (const float*)d_in[10];
    (void)in_sizes; (void)n_in; (void)out_size; (void)ws_size;

    // workspace carve-up (256B aligned)
    uint8_t* base = (uint8_t*)d_ws;
    size_t off = 0;
    auto take = [&](size_t bytes) -> void* {
        void* p = base + off;
        off += (bytes + 255) & ~(size_t)255;
        return p;
    };
    _Float16* wt1 = (_Float16*)take((size_t)NG * K1 * sizeof(_Float16));   // 2 MB
    _Float16* wt2 = (_Float16*)take((size_t)NG * K2 * sizeof(_Float16));   // 4 MB
    _Float16* h1  = (_Float16*)take((size_t)2 * BATCH * HID * sizeof(_Float16));
    _Float16* h2  = (_Float16*)take((size_t)2 * BATCH * HID * sizeof(_Float16));
    float*    b1  = (float*)take(NG * sizeof(float));
    float*    b2  = (float*)take(NG * sizeof(float));
    unsigned* bar = (unsigned*)take(256);

    init_state_kernel<<<512, 256, 0, stream>>>(h1, h2, b1, b2,
                                               b_ih1, b_hh1, b_ih2, b_hh2, bar);
    convert_weights_kernel<<<4096, 256, 0, stream>>>(w_hh1, w_ih2, w_hh2, wt1, wt2);

    (void)hipFuncSetAttribute(reinterpret_cast<const void*>(&lstm_persistent),
                              hipFuncAttributeMaxDynamicSharedMemorySize, LDS_BYTES);
    lstm_persistent<<<NWGS, THREADS, LDS_BYTES, stream>>>(x, w_ih1, b1, b2, wt1, wt2,
                                                          h1, h2, bar);

    // final h2 is in buffer 0: t=511 -> rb=1, wb=0
    fc_softmax_kernel<<<1, 256, 0, stream>>>(h2, fc_w, fc_b, (float*)d_out);
}
